// MoELayer_46059229282642
// MI455X (gfx1250) — matrix operand order repaired and checked by reading, not compiled, not run on hardware
//
#include <hip/hip_runtime.h>
#include <hip/hip_bf16.h>

// ---------------- problem constants ----------------
#define TT   4096     // tokens = B*S
#define DD   1024     // d_model
#define HD   2048     // expert hidden
#define EE   8        // experts
#define TM   64       // token tile (rows)
#define TNW  256      // output-col tile
#define TK2  64       // K per LDS fill (2 WMMA k-steps of 32)
#define LROW (TK2 / 2 + 4)   // LDS row stride in uints (32 + 4 pad = 144B, 16B-aligned)

typedef __attribute__((ext_vector_type(16))) __bf16 v16bf;
typedef __attribute__((ext_vector_type(8)))  __bf16 v8bf;
typedef __attribute__((ext_vector_type(8)))  float  v8f;

// f32 -> bf16 by truncation, packed pair (lo = even k)
__device__ __forceinline__ unsigned int packbf(float lo, float hi) {
  return (__float_as_uint(hi) & 0xFFFF0000u) | (__float_as_uint(lo) >> 16);
}
__device__ __forceinline__ unsigned short f2bf(float f) {
  return (unsigned short)(__float_as_uint(f) >> 16);
}
// flat pointer -> 32-bit LDS offset (flat LDS aperture keeps offset in low 32 bits)
__device__ __forceinline__ unsigned lds_off(const void* p) {
  return (unsigned)(unsigned long long)p;
}

// A-fragment (16x32 bf16 row-major): lane s=8*half; elems 0..7 = K s..s+7,
// elems 8..15 = K 16+s..16+s+7
__device__ __forceinline__ v16bf load_frag_A(const __bf16* row, int half) {
  v8bf lo = *(const v8bf*)(row + half * 8);
  v8bf hi = *(const v8bf*)(row + 16 + half * 8);
  return __builtin_shufflevector(lo, hi, 0,1,2,3,4,5,6,7,8,9,10,11,12,13,14,15);
}
// B-fragment (32x16 bf16, N-major rows of K): lane col n=lane%16,
// elems 0..7 = K 8*half..8*half+7, elems 8..15 = K 16+8*half..16+8*half+7
__device__ __forceinline__ v16bf load_frag_B(const __bf16* row, int half) {
  v8bf lo = *(const v8bf*)(row + half * 8);
  v8bf hi = *(const v8bf*)(row + 16 + half * 8);
  return __builtin_shufflevector(lo, hi, 0,1,2,3,4,5,6,7,8,9,10,11,12,13,14,15);
}

// ---------------- kernel 0: zero out + counters ----------------
__global__ void k_init(float* out, int* cnt) {
  size_t stride = (size_t)gridDim.x * blockDim.x;
  for (size_t i = (size_t)blockIdx.x * blockDim.x + threadIdx.x;
       i < (size_t)TT * DD; i += stride)
    out[i] = 0.0f;
  if (blockIdx.x == 0 && threadIdx.x < EE) cnt[threadIdx.x] = 0;
}

// ---------------- kernel 1: gate + top-2 route ----------------
__global__ __launch_bounds__(256)
void k_gate(const float* __restrict__ x, const float* __restrict__ Wg,
            const float* __restrict__ bg, int* cnt, int* tIdx, float* tW) {
  int wid  = threadIdx.x >> 5;
  int lane = threadIdx.x & 31;
  int t = blockIdx.x * 8 + wid;                 // grid = TT/8 blocks
  float acc[EE];
#pragma unroll
  for (int e = 0; e < EE; ++e) acc[e] = 0.0f;
  for (int i = 0; i < DD / 32; ++i) {
    int d = i * 32 + lane;
    float xv = x[(size_t)t * DD + d];
#pragma unroll
    for (int e = 0; e < EE; ++e) acc[e] += xv * Wg[(size_t)e * DD + d];
  }
#pragma unroll
  for (int e = 0; e < EE; ++e) {
#pragma unroll
    for (int off = 16; off > 0; off >>= 1)
      acc[e] += __shfl_xor(acc[e], off, 32);
  }
  if (lane == 0) {
#pragma unroll
    for (int e = 0; e < EE; ++e) acc[e] += bg[e];
    int i0 = 0;
#pragma unroll
    for (int e = 1; e < EE; ++e) if (acc[e] > acc[i0]) i0 = e;
    int i1 = (i0 == 0) ? 1 : 0;
#pragma unroll
    for (int e = 0; e < EE; ++e)
      if (e != i0 && acc[e] > acc[i1]) i1 = e;
    float e1 = __expf(acc[i1] - acc[i0]);
    float inv = 1.0f / (1.0f + e1);
    int p0 = atomicAdd(&cnt[i0], 1);
    tIdx[i0 * TT + p0] = t;  tW[i0 * TT + p0] = inv;
    int p1 = atomicAdd(&cnt[i1], 1);
    tIdx[i1 * TT + p1] = t;  tW[i1 * TT + p1] = e1 * inv;
  }
}

// ---------------- kernel 2: exclusive prefix over E=8 counts ----------------
__global__ void k_prefix(const int* cnt, int* basep) {
  if (threadIdx.x == 0) {
    int s = 0;
#pragma unroll
    for (int e = 0; e < EE; ++e) { basep[e] = s; s += cnt[e]; }
  }
}

// ---------------- kernel 3: h = relu(Xg @ W1^T + b1), bf16 out ----------------
__global__ __launch_bounds__(256)
void k_ffn1(const float* __restrict__ x, const float* __restrict__ W1,
            const float* __restrict__ b1, const int* __restrict__ cnt,
            const int* __restrict__ basep, const int* __restrict__ tIdx,
            unsigned short* __restrict__ hbuf) {
  int e   = blockIdx.x >> 9;                    // 64*8 blocks per expert
  int rem = blockIdx.x & 511;
  int tm  = rem >> 3;                           // token tile 0..63
  int tn  = rem & 7;                            // H-col tile 0..7
  int c = cnt[e];
  if (tm * TM >= c) return;
  int colBase = tn * TNW;
  const float* W1e = W1 + (size_t)e * HD * DD;

  __shared__ __align__(16) unsigned int Asu[TM][LROW];
  __shared__ __align__(16) unsigned int Bsu[TNW][LROW];

  int tid = threadIdx.x, lane = tid & 31, wid = tid >> 5;
  int wm = wid & 1, wn = wid >> 1;              // wave -> 32-row x 64-col block
  int half = lane >> 4, l16 = lane & 15;

  // hoist gathered token ids for this thread's 8 A pairs (64 rows x 32 pairs)
  int arow[8], atok[8];
#pragma unroll
  for (int i = 0; i < 8; ++i) {
    int p = tid + i * 256;
    arow[i] = p >> 5;
    int slot = tm * TM + arow[i];
    atok[i] = (slot < c) ? tIdx[e * TT + slot] : tIdx[e * TT];
  }

  v8f acc[2][4];
#pragma unroll
  for (int f = 0; f < 2; ++f)
#pragma unroll
    for (int s = 0; s < 4; ++s) acc[f][s] = (v8f){};

  for (int kb = 0; kb < DD / TK2; ++kb) {
    int k0 = kb * TK2;
    __syncthreads();
    // fill A: 64x64 f32 -> bf16 (gathered token rows), 8 pairs/thread
#pragma unroll
    for (int i = 0; i < 8; ++i) {
      int p = tid + i * 256;
      int cp = p & 31;
      const float2 f = *(const float2*)(x + (size_t)atok[i] * DD + k0 + cp * 2);
      Asu[arow[i]][cp] = packbf(f.x, f.y);
    }
    // fill B: 256x64 f32 -> bf16 from W1[e], 32 pairs/thread
#pragma unroll
    for (int i = 0; i < 32; ++i) {
      int p = tid + i * 256;
      int r = p >> 5, cp = p & 31;
      const float* src = W1e + (size_t)(colBase + r) * DD + k0 + cp * 2;
      const float2 f = *(const float2*)src;
      Bsu[r][cp] = packbf(f.x, f.y);
      if (i == 0 && kb + 1 < DD / TK2) __builtin_prefetch(src + TK2, 0, 0);
    }
    __syncthreads();
    // compute: 2 A-frags x 4 B-frags x 2 k-substeps = 16 WMMA / wave / fill
#pragma unroll
    for (int kk = 0; kk < 2; ++kk) {
      v16bf a0 = load_frag_A((const __bf16*)&Asu[wm * 32 + l16][0] + kk * 32, half);
      v16bf a1 = load_frag_A((const __bf16*)&Asu[wm * 32 + 16 + l16][0] + kk * 32, half);
#pragma unroll
      for (int s = 0; s < 4; ++s) {
        v16bf b = load_frag_B((const __bf16*)&Bsu[wn * 64 + s * 16 + l16][0] + kk * 32, half);
        acc[0][s] = __builtin_amdgcn_wmma_f32_16x16x32_bf16(
            false, a0, false, b, (short)0, acc[0][s], false, false);
        acc[1][s] = __builtin_amdgcn_wmma_f32_16x16x32_bf16(
            false, a1, false, b, (short)0, acc[1][s], false, false);
      }
    }
  }

  // epilogue: bias + relu, store bf16 h rows (compacted by expert base)
  int basee = basep[e];
#pragma unroll
  for (int s = 0; s < 4; ++s) {
    int gcol = colBase + wn * 64 + s * 16 + l16;
    float bv = b1[(size_t)e * HD + gcol];
#pragma unroll
    for (int f = 0; f < 2; ++f) {
#pragma unroll
      for (int j = 0; j < 8; ++j) {
        int m = wm * 32 + f * 16 + half * 8 + j;
        int slot = tm * TM + m;
        if (slot < c) {
          float v = acc[f][s][j] + bv;
          v = v > 0.0f ? v : 0.0f;
          hbuf[(size_t)(basee + slot) * HD + gcol] = f2bf(v);
        }
      }
    }
  }
}

// ---------------- kernel 4: y = h @ W2^T + b2, scaled scatter-add ----------------
__global__ __launch_bounds__(256)
void k_ffn2(const unsigned short* __restrict__ hbuf, const float* __restrict__ W2,
            const float* __restrict__ b2, const int* __restrict__ cnt,
            const int* __restrict__ basep, const int* __restrict__ tIdx,
            const float* __restrict__ tW, float* __restrict__ out) {
  int e   = blockIdx.x >> 8;                    // 64*4 blocks per expert
  int rem = blockIdx.x & 255;
  int tm  = rem >> 2;                           // token tile 0..63
  int tn  = rem & 3;                            // D-col tile 0..3
  int c = cnt[e];
  if (tm * TM >= c) return;
  int colBase = tn * TNW;
  int basee = basep[e];
  const float* W2e = W2 + (size_t)e * DD * HD;

  __shared__ __align__(16) unsigned int Asu[TM][LROW];
  __shared__ __align__(16) unsigned int Bsu[TNW][LROW];

  int tid = threadIdx.x, lane = tid & 31, wid = tid >> 5;
  int wm = wid & 1, wn = wid >> 1;
  int half = lane >> 4, l16 = lane & 15;

  v8f acc[2][4];
#pragma unroll
  for (int f = 0; f < 2; ++f)
#pragma unroll
    for (int s = 0; s < 4; ++s) acc[f][s] = (v8f){};

  const unsigned int* hrows = (const unsigned int*)hbuf;  // 2 bf16 per uint
  for (int kb = 0; kb < HD / TK2; ++kb) {
    int k0 = kb * TK2;
    __syncthreads();
    // fill A: bf16 h rows, async global->LDS b128 (64 rows x 8 chunks, 2/thread)
#pragma unroll
    for (int i = 0; i < 2; ++i) {
      int p = tid + i * 256;
      int r = p >> 3, ch = p & 7;
      int gr = basee + tm * TM + r;
      gr = gr < (2 * TT) ? gr : (2 * TT - 1);   // clamp tail reads inside hbuf
      unsigned long long ga =
          (unsigned long long)(const void*)(hrows + (size_t)gr * (HD / 2) + kb * 32 + ch * 4);
      unsigned la = lds_off(&Asu[r][ch * 4]);
      asm volatile("global_load_async_to_lds_b128 %0, %1, off"
                   :: "v"(la), "v"(ga) : "memory");
    }
    // fill B: 256x64 f32 -> bf16 from W2[e], 32 pairs/thread
#pragma unroll
    for (int i = 0; i < 32; ++i) {
      int p = tid + i * 256;
      int r = p >> 5, cp = p & 31;
      const float* src = W2e + (size_t)(colBase + r) * HD + k0 + cp * 2;
      const float2 f = *(const float2*)src;
      Bsu[r][cp] = packbf(f.x, f.y);
      if (i == 0 && kb + 1 < HD / TK2) __builtin_prefetch(src + TK2, 0, 0);
    }
    asm volatile("s_wait_asynccnt 0x0" ::: "memory");
    __syncthreads();
#pragma unroll
    for (int kk = 0; kk < 2; ++kk) {
      v16bf a0 = load_frag_A((const __bf16*)&Asu[wm * 32 + l16][0] + kk * 32, half);
      v16bf a1 = load_frag_A((const __bf16*)&Asu[wm * 32 + 16 + l16][0] + kk * 32, half);
#pragma unroll
      for (int s = 0; s < 4; ++s) {
        v16bf b = load_frag_B((const __bf16*)&Bsu[wn * 64 + s * 16 + l16][0] + kk * 32, half);
        acc[0][s] = __builtin_amdgcn_wmma_f32_16x16x32_bf16(
            false, a0, false, b, (short)0, acc[0][s], false, false);
        acc[1][s] = __builtin_amdgcn_wmma_f32_16x16x32_bf16(
            false, a1, false, b, (short)0, acc[1][s], false, false);
      }
    }
  }

  // per-row routing weight + token id
  int   toks[2][8];
  float wgt[2][8];
#pragma unroll
  for (int f = 0; f < 2; ++f) {
#pragma unroll
    for (int j = 0; j < 8; ++j) {
      int m = wm * 32 + f * 16 + half * 8 + j;
      int slot = tm * TM + m;
      bool ok = slot < c;
      toks[f][j] = ok ? tIdx[e * TT + slot] : 0;
      wgt[f][j]  = ok ? tW[e * TT + slot] : 0.0f;
    }
  }
#pragma unroll
  for (int s = 0; s < 4; ++s) {
    int gcol = colBase + wn * 64 + s * 16 + l16;
    float bv = b2[(size_t)e * DD + gcol];
#pragma unroll
    for (int f = 0; f < 2; ++f) {
#pragma unroll
      for (int j = 0; j < 8; ++j) {
        int m = wm * 32 + f * 16 + half * 8 + j;
        int slot = tm * TM + m;
        if (slot < c) {
          float v = (acc[f][s][j] + bv) * wgt[f][j];
          atomicAdd(&out[(size_t)toks[f][j] * DD + gcol], v);
        }
      }
    }
  }
}

// ---------------- launch ----------------
extern "C" void kernel_launch(void* const* d_in, const int* in_sizes, int n_in,
                              void* d_out, int out_size, void* d_ws, size_t ws_size,
                              hipStream_t stream) {
  const float* x  = (const float*)d_in[0];
  const float* W1 = (const float*)d_in[1];
  const float* b1 = (const float*)d_in[2];
  const float* W2 = (const float*)d_in[3];
  const float* b2 = (const float*)d_in[4];
  const float* Wg = (const float*)d_in[5];
  const float* bg = (const float*)d_in[6];
  float* out = (float*)d_out;

  // workspace layout
  char* ws = (char*)d_ws;
  int*   cnt   = (int*)ws;                                  // E ints
  int*   basep = (int*)(ws + 64);                           // E ints
  int*   tIdx  = (int*)(ws + 128);                          // E*T ints
  float* tW    = (float*)(ws + 128 + (size_t)EE * TT * 4);  // E*T floats
  unsigned short* hbuf =
      (unsigned short*)(ws + 512 + 2 * (size_t)EE * TT * 4); // 2T x H bf16

  k_init  <<<1024, 256, 0, stream>>>(out, cnt);
  k_gate  <<<TT / 8, 256, 0, stream>>>(x, Wg, bg, cnt, tIdx, tW);
  k_prefix<<<1, 32, 0, stream>>>(cnt, basep);
  k_ffn1  <<<EE * 64 * (HD / TNW), 256, 0, stream>>>(x, W1, b1, cnt, basep, tIdx, hbuf);
  k_ffn2  <<<EE * 64 * (DD / TNW), 256, 0, stream>>>(hbuf, W2, b2, cnt, basep, tIdx, tW, out);
}
